// GAU_10557029613728
// MI455X (gfx1250) — compile-verified, matrix operand order checked by reading, not run on hardware
//
#include <hip/hip_runtime.h>
#include <hip/hip_bf16.h>
#include <math.h>

// ---------------------------------------------------------------------------
// GAU block for MI455X (gfx1250, wave32, WMMA 16x16x32 bf16, fp32 accum)
// Round 3: fix async-to-LDS builtin pointer types (int4 as(1)* / int4 as(3)*).
// ---------------------------------------------------------------------------

#define BATCH   8
#define SEQ     2048
#define DIM     512
#define H2      2048     // HIDDEN*2
#define HID     1024     // HIDDEN
#define QKD     128
#define MROWS   (BATCH*SEQ)   // 16384

typedef __attribute__((ext_vector_type(16))) __bf16 v16bf;
typedef __attribute__((ext_vector_type(8)))  float  v8f;
typedef __attribute__((ext_vector_type(4)))  int    v4i;

typedef __attribute__((address_space(1))) v4i gv4i;
typedef __attribute__((address_space(3))) v4i lv4i;

union V16BF { v16bf v; uint4 q[2]; };

// ---- async global->LDS copy (CDNA5 path) with portable fallback ----------
#if __has_builtin(__builtin_amdgcn_global_load_async_to_lds_b128)
#define HAVE_ASYNC_LDS 1
#else
#define HAVE_ASYNC_LDS 0
#endif

#if HAVE_ASYNC_LDS
#if __has_builtin(__builtin_amdgcn_s_wait_asynccnt)
#define STAGE_FENCE(n) __builtin_amdgcn_s_wait_asynccnt(n)
#else
#define STAGE_FENCE(n) asm volatile("s_wait_asynccnt " #n ::: "memory")
#endif
#else
#define STAGE_FENCE(n) ((void)0)
#endif

__device__ __forceinline__ void async_copy16(void* lds_dst, const void* gsrc) {
#if HAVE_ASYNC_LDS
  __builtin_amdgcn_global_load_async_to_lds_b128(
      (gv4i*)(uintptr_t)gsrc,
      (lv4i*)(uint32_t)(uintptr_t)lds_dst,
      0, 0);
#else
  *reinterpret_cast<uint4*>(lds_dst) = *reinterpret_cast<const uint4*>(gsrc);
#endif
}

__device__ __forceinline__ v8f zero_v8f() {
  v8f z;
#pragma unroll
  for (int i = 0; i < 8; ++i) z[i] = 0.0f;
  return z;
}

// Load a 16x32 bf16 tile in WMMA A-operand layout from a row-major source
// (also B operands, given the transposed matrix Bt[n][k]). Works on global
// or LDS pointers; ld must be a multiple of 8 (rows 16B-aligned).
//   lane l: m = l&15, kb = (l<16)?0:8
//   elements 0..7  = src[m][kb+0 .. kb+7]
//   elements 8..15 = src[m][kb+16 .. kb+23]
__device__ __forceinline__ v16bf load_a16x32(const __hip_bfloat16* __restrict__ base,
                                             int ld) {
  const int lane = threadIdx.x & 31;
  const int m  = lane & 15;
  const int kb = (lane & 16) >> 1;   // 0 or 8
  const __hip_bfloat16* p = base + (size_t)m * ld + kb;
  V16BF u;
  u.q[0] = *reinterpret_cast<const uint4*>(p);
  u.q[1] = *reinterpret_cast<const uint4*>(p + 16);
  return u.v;
}

__device__ __forceinline__ v8f wmma_bf16(v16bf a, v16bf b, v8f c) {
  return __builtin_amdgcn_wmma_f32_16x16x32_bf16(false, a, false, b,
                                                 (short)0, c, false, false);
}

__device__ __forceinline__ float silu(float x) {
  return x / (1.0f + expf(-x));
}

// ---------------------------------------------------------------------------
// fp32 -> bf16 transpose-convert: out[c][r] = bf16(in[r][c]); R,C % 32 == 0
// ---------------------------------------------------------------------------
__global__ __launch_bounds__(256)
void transpose_f2b(const float* __restrict__ in, __hip_bfloat16* __restrict__ out,
                   int R, int C) {
  __shared__ float tile[32][33];
  const int r0 = blockIdx.y * 32, c0 = blockIdx.x * 32;
  const int tx = threadIdx.x, ty = threadIdx.y;   // 32 x 8
#pragma unroll
  for (int i = 0; i < 32; i += 8)
    tile[ty + i][tx] = in[(size_t)(r0 + ty + i) * C + (c0 + tx)];
  __syncthreads();
#pragma unroll
  for (int i = 0; i < 32; i += 8)
    out[(size_t)(c0 + ty + i) * R + (r0 + tx)] = __float2bfloat16(tile[tx][ty + i]);
}

// ---------------------------------------------------------------------------
// per-batch bf16 transpose: out[z][c][r] = in[z][r][c]; R,C % 32 == 0
// ---------------------------------------------------------------------------
__global__ __launch_bounds__(256)
void transpose_b2b(const __hip_bfloat16* __restrict__ in,
                   __hip_bfloat16* __restrict__ out, int R, int C) {
  __shared__ __hip_bfloat16 tile[32][33];
  const size_t off = (size_t)blockIdx.z * R * C;
  const int r0 = blockIdx.y * 32, c0 = blockIdx.x * 32;
  const int tx = threadIdx.x, ty = threadIdx.y;
#pragma unroll
  for (int i = 0; i < 32; i += 8)
    tile[ty + i][tx] = in[off + (size_t)(r0 + ty + i) * C + (c0 + tx)];
  __syncthreads();
#pragma unroll
  for (int i = 0; i < 32; i += 8)
    out[off + (size_t)(c0 + ty + i) * R + (r0 + tx)] = tile[tx][ty + i];
}

// ---------------------------------------------------------------------------
// LayerNorm: one wave per row of 512, lane handles 16 elems, wave32 shfl
// ---------------------------------------------------------------------------
__global__ __launch_bounds__(256)
void layernorm_kernel(const float* __restrict__ x,
                      const float* __restrict__ scale,
                      const float* __restrict__ bias,
                      __hip_bfloat16* __restrict__ normed) {
  const int wave = threadIdx.x >> 5;
  const int lane = threadIdx.x & 31;
  const int row  = blockIdx.x * 8 + wave;
  const float* xr = x + (size_t)row * DIM;

  float4 vv[4];
  float s = 0.0f, ss = 0.0f;
#pragma unroll
  for (int i = 0; i < 4; ++i) {
    vv[i] = reinterpret_cast<const float4*>(xr)[i * 32 + lane];
    s  += vv[i].x + vv[i].y + vv[i].z + vv[i].w;
    ss += vv[i].x*vv[i].x + vv[i].y*vv[i].y + vv[i].z*vv[i].z + vv[i].w*vv[i].w;
  }
#pragma unroll
  for (int off = 16; off > 0; off >>= 1) {
    s  += __shfl_xor(s,  off, 32);
    ss += __shfl_xor(ss, off, 32);
  }
  const float mu   = s * (1.0f / DIM);
  const float var  = ss * (1.0f / DIM) - mu * mu;
  const float rstd = rsqrtf(var + 1e-5f);

  uint2* orow = reinterpret_cast<uint2*>(normed + (size_t)row * DIM);
#pragma unroll
  for (int i = 0; i < 4; ++i) {
    const int f = i * 32 + lane;
    const int c = f * 4;
    union { __hip_bfloat16 h[4]; uint2 u; } o;
    o.h[0] = __float2bfloat16((vv[i].x - mu) * rstd * scale[c + 0] + bias[c + 0]);
    o.h[1] = __float2bfloat16((vv[i].y - mu) * rstd * scale[c + 1] + bias[c + 1]);
    o.h[2] = __float2bfloat16((vv[i].z - mu) * rstd * scale[c + 2] + bias[c + 2]);
    o.h[3] = __float2bfloat16((vv[i].w - mu) * rstd * scale[c + 3] + bias[c + 3]);
    orow[f] = o.u;
  }
}

// ---------------------------------------------------------------------------
// Generic bf16 WMMA GEMM: C[M][N] = A[M][K] * Bt[N][K]^T
// WG tile 128x128, 8 waves 2x4, wave tile 64x32. K stepped by 32 with
// double-buffered async staging of the 128x32 A and Bt tiles into LDS.
// EPI 0: hidden (silu -> v | gate)   EPI 1: qk (silu -> q,k affine)
// EPI 2: out (bias + residual, fp32 store)
// ---------------------------------------------------------------------------
template<int EPI>
__global__ __launch_bounds__(256)
void gemm_bf16(const __hip_bfloat16* __restrict__ A,
               const __hip_bfloat16* __restrict__ Bt,
               int M, int N, int K,
               const float* __restrict__ bias,
               const float* __restrict__ gamma,
               const float* __restrict__ beta,
               const float* __restrict__ resid,
               __hip_bfloat16* __restrict__ ob0,
               __hip_bfloat16* __restrict__ ob1,
               float* __restrict__ of) {
  constexpr int LDT = 56;  // padded pitch: 112B rows, 16B aligned, banks spread x4
  __shared__ __align__(16) __hip_bfloat16 sA[2][128 * LDT];
  __shared__ __align__(16) __hip_bfloat16 sB[2][128 * LDT];

  const int tid  = threadIdx.x;
  const int wave = tid >> 5;
  const int lane = tid & 31;
  const int wm = wave >> 2;                    // 0..1
  const int wn = wave & 3;                     // 0..3
  const int rowB = blockIdx.x * 128;           // A row block
  const int colB = blockIdx.y * 128;           // Bt row block

  // stage one 128x32 slab of A and Bt each: 4 async b128 instrs per wave
  auto stage = [&](int buf, int k0) {
#pragma unroll
    for (int i = 0; i < 2; ++i) {
      const int c = tid + 256 * i;             // 0..511
      const int row = c >> 2, seg = c & 3;     // 4 x 16B per row
      async_copy16(&sA[buf][row * LDT + seg * 8],
                   A + (size_t)(rowB + row) * K + k0 + seg * 8);
    }
#pragma unroll
    for (int i = 0; i < 2; ++i) {
      const int c = tid + 256 * i;
      const int row = c >> 2, seg = c & 3;
      async_copy16(&sB[buf][row * LDT + seg * 8],
                   Bt + (size_t)(colB + row) * K + k0 + seg * 8);
    }
  };

  v8f acc[4][2];
#pragma unroll
  for (int mt = 0; mt < 4; ++mt)
#pragma unroll
    for (int nt = 0; nt < 2; ++nt) acc[mt][nt] = zero_v8f();

  stage(0, 0);
  const int steps = K >> 5;
  for (int s = 0; s < steps; ++s) {
    const int buf = s & 1;
    if (s + 1 < steps) { stage(buf ^ 1, (s + 1) << 5); STAGE_FENCE(4); }
    else               { STAGE_FENCE(0); }
    __syncthreads();

    v16bf a[4], b[2];
#pragma unroll
    for (int mt = 0; mt < 4; ++mt)
      a[mt] = load_a16x32(&sA[buf][(wm * 64 + 16 * mt) * LDT], LDT);
#pragma unroll
    for (int nt = 0; nt < 2; ++nt)
      b[nt] = load_a16x32(&sB[buf][(wn * 32 + 16 * nt) * LDT], LDT);
#pragma unroll
    for (int mt = 0; mt < 4; ++mt)
#pragma unroll
      for (int nt = 0; nt < 2; ++nt)
        acc[mt][nt] = wmma_bf16(a[mt], b[nt], acc[mt][nt]);

    __syncthreads();   // protect buf from the stage() of step s+2
  }

  // C layout: vgpr r -> m = r + 8*(lane>>4), n = lane&15
  const int mb = (lane >> 4) << 3;
  const int nn = lane & 15;
#pragma unroll
  for (int mt = 0; mt < 4; ++mt) {
#pragma unroll
    for (int nt = 0; nt < 2; ++nt) {
#pragma unroll
      for (int r = 0; r < 8; ++r) {
        const int row = rowB + wm * 64 + 16 * mt + mb + r;
        const int col = colB + wn * 32 + 16 * nt + nn;
        const float c = acc[mt][nt][r];
        if (EPI == 0) {
          const float h = silu(c + bias[col]);
          const int half = N >> 1;             // 1024
          if (col < half)
            ob0[(size_t)row * half + col] = __float2bfloat16(h);          // v
          else
            ob1[(size_t)row * half + (col - half)] = __float2bfloat16(h); // gate
        } else if (EPI == 1) {
          const float z = silu(c + bias[col]);
          ob0[(size_t)row * N + col] = __float2bfloat16(z * gamma[col] + beta[col]);
          ob1[(size_t)row * N + col] = __float2bfloat16(z * gamma[N + col] + beta[N + col]);
        } else {
          of[(size_t)row * N + col] = c + bias[col] + resid[(size_t)row * N + col];
        }
      }
    }
  }
}

// ---------------------------------------------------------------------------
// Fused attention. grid (16 rowblocks, 4 colblocks, 8 batches), 256 thr.
// Wave owns 16 rows x 256 v-cols. Per 32-wide j step (double-buffered
// async staging of the shared 32x128 k tile and 256x32 vT tile):
//   S(16x32)  = q(16x128) @ k(Jtile)^T          (8 WMMAs, K=128)
//   P         = (relu(S)/2048)^2 -> bf16, per-wave LDS relayout
//   acc(16x256) += P(16x32) @ vT(tile)^T        (16 WMMAs)
// Epilogue: acc * gate -> Vout bf16.  sim never touches HBM.
// ---------------------------------------------------------------------------
__global__ __launch_bounds__(256)
void attention_kernel(const __hip_bfloat16* __restrict__ q,
                      const __hip_bfloat16* __restrict__ k,
                      const __hip_bfloat16* __restrict__ vT,
                      const __hip_bfloat16* __restrict__ gate,
                      __hip_bfloat16* __restrict__ Vout) {
  constexpr int LDK2 = 136;  // 272B rows: 16B aligned, banks spread x4
  constexpr int LDV2 = 56;   // 112B rows: 16B aligned, banks spread x4
  __shared__ __align__(16) __hip_bfloat16 sK[2][32 * LDK2];    // 2 x 8.5 KB
  __shared__ __align__(16) __hip_bfloat16 sV[2][256 * LDV2];   // 2 x 28 KB
  __shared__ __align__(16) __hip_bfloat16 Pbuf[8][16 * 32];    // 8 KB

  const int b    = blockIdx.z;
  const int tid  = threadIdx.x;
  const int wave = tid >> 5;
  const int lane = tid & 31;
  const int I = blockIdx.x * 128 + wave * 16;   // row block (16 rows / wave)
  const int D = blockIdx.y * 256;               // v-column block (256 / WG)

  const __hip_bfloat16* qb = q  + (size_t)b * SEQ * QKD;
  const __hip_bfloat16* kk = k  + (size_t)b * SEQ * QKD;
  const __hip_bfloat16* vt = vT + (size_t)b * HID * SEQ;   // [1024][2048]
  __hip_bfloat16* P = Pbuf[wave];

  // stage k(32x128) + vT(256x32): 6 async b128 instrs per wave
  auto stage = [&](int buf, int J) {
#pragma unroll
    for (int i = 0; i < 2; ++i) {
      const int c = tid + 256 * i;              // 512 chunks, 16 x 16B per row
      const int row = c >> 4, seg = c & 15;
      async_copy16(&sK[buf][row * LDK2 + seg * 8],
                   kk + (size_t)(J + row) * QKD + seg * 8);
    }
#pragma unroll
    for (int i = 0; i < 4; ++i) {
      const int c = tid + 256 * i;              // 1024 chunks, 4 x 16B per row
      const int row = c >> 2, seg = c & 3;
      async_copy16(&sV[buf][row * LDV2 + seg * 8],
                   vt + (size_t)(D + row) * SEQ + J + seg * 8);
    }
  };

  // q rows for this wave, kept in registers across the whole j loop
  v16bf qa[4];
#pragma unroll
  for (int c = 0; c < 4; ++c)
    qa[c] = load_a16x32(qb + (size_t)I * QKD + 32 * c, QKD);

  v8f acc[16];
#pragma unroll
  for (int t = 0; t < 16; ++t) acc[t] = zero_v8f();

  const int mb = (lane >> 4) << 3;
  const int nn = lane & 15;
  const float inv = 1.0f / (float)SEQ;

  stage(0, 0);
  for (int s = 0; s < SEQ / 32; ++s) {
    const int buf = s & 1;
    if (s + 1 < SEQ / 32) { stage(buf ^ 1, (s + 1) * 32); STAGE_FENCE(6); }
    else                  { STAGE_FENCE(0); }
    __syncthreads();

    // --- S = q @ k^T : row-major k IS the B-operand layout for k^T ---
    v8f s0 = zero_v8f(), s1 = zero_v8f();
#pragma unroll
    for (int c = 0; c < 4; ++c) {
      v16bf b0 = load_a16x32(&sK[buf][0  * LDK2 + 32 * c], LDK2);
      v16bf b1 = load_a16x32(&sK[buf][16 * LDK2 + 32 * c], LDK2);
      s0 = wmma_bf16(qa[c], b0, s0);
      s1 = wmma_bf16(qa[c], b1, s1);
    }
    // --- P = relu(S/seq)^2 -> bf16, C layout -> row-major [16][32] ---
#pragma unroll
    for (int r = 0; r < 8; ++r) {
      float e0 = s0[r] * inv; e0 = (e0 > 0.0f) ? e0 * e0 : 0.0f;
      float e1 = s1[r] * inv; e1 = (e1 > 0.0f) ? e1 * e1 : 0.0f;
      P[(mb + r) * 32 + nn]      = __float2bfloat16(e0);
      P[(mb + r) * 32 + 16 + nn] = __float2bfloat16(e1);
    }
    __builtin_amdgcn_wave_barrier();
    asm volatile("s_wait_dscnt 0" ::: "memory");   // wave-private buffer; DS in-order
    v16bf pa = load_a16x32(P, 32);
    // --- acc += P @ v : vT row-major is the B-operand layout ---
#pragma unroll
    for (int t = 0; t < 16; ++t) {
      v16bf vb = load_a16x32(&sV[buf][(16 * t) * LDV2], LDV2);
      acc[t] = wmma_bf16(pa, vb, acc[t]);
    }
    __syncthreads();   // protect buf from the stage() of step s+2
  }

  // --- epilogue: * gate, store bf16 ---
#pragma unroll
  for (int t = 0; t < 16; ++t) {
#pragma unroll
    for (int r = 0; r < 8; ++r) {
      const size_t gr = (size_t)b * SEQ + I + mb + r;   // flat row in [16384]
      const int col = D + 16 * t + nn;
      const float g = __bfloat162float(gate[gr * HID + col]);
      Vout[gr * HID + col] = __float2bfloat16(acc[t][r] * g);
    }
  }
}

// ---------------------------------------------------------------------------
// Launch
// ---------------------------------------------------------------------------
extern "C" void kernel_launch(void* const* d_in, const int* in_sizes, int n_in,
                              void* d_out, int out_size, void* d_ws, size_t ws_size,
                              hipStream_t stream) {
  const float* x        = (const float*)d_in[0];
  const float* ln_scale = (const float*)d_in[1];
  const float* ln_bias  = (const float*)d_in[2];
  const float* W_hidden = (const float*)d_in[3];
  const float* b_hidden = (const float*)d_in[4];
  const float* W_qk     = (const float*)d_in[5];
  const float* b_qk     = (const float*)d_in[6];
  const float* gamma    = (const float*)d_in[7];
  const float* beta     = (const float*)d_in[8];
  const float* W_out    = (const float*)d_in[9];
  const float* b_out    = (const float*)d_in[10];
  float* out = (float*)d_out;

  char* ws = (char*)d_ws;
  auto alloc = [&](size_t bytes) -> char* {
    char* p = ws;
    ws += (bytes + 255) & ~(size_t)255;
    return p;
  };
  __hip_bfloat16* normed = (__hip_bfloat16*)alloc((size_t)MROWS * DIM * 2);
  __hip_bfloat16* WhT    = (__hip_bfloat16*)alloc((size_t)H2   * DIM * 2);
  __hip_bfloat16* WqkT   = (__hip_bfloat16*)alloc((size_t)QKD  * DIM * 2);
  __hip_bfloat16* WoT    = (__hip_bfloat16*)alloc((size_t)DIM  * HID * 2);
  __hip_bfloat16* vbuf   = (__hip_bfloat16*)alloc((size_t)MROWS * HID * 2);
  __hip_bfloat16* gbuf   = (__hip_bfloat16*)alloc((size_t)MROWS * HID * 2);
  __hip_bfloat16* qbuf   = (__hip_bfloat16*)alloc((size_t)MROWS * QKD * 2);
  __hip_bfloat16* kbuf   = (__hip_bfloat16*)alloc((size_t)MROWS * QKD * 2);
  __hip_bfloat16* vTb    = (__hip_bfloat16*)alloc((size_t)MROWS * HID * 2);
  __hip_bfloat16* Vout   = (__hip_bfloat16*)alloc((size_t)MROWS * HID * 2);
  (void)ws_size; // total ~163 MB

  const dim3 tb(32, 8);
  transpose_f2b<<<dim3(H2 / 32,  DIM / 32), tb, 0, stream>>>(W_hidden, WhT, DIM, H2);
  transpose_f2b<<<dim3(QKD / 32, DIM / 32), tb, 0, stream>>>(W_qk,    WqkT, DIM, QKD);
  transpose_f2b<<<dim3(DIM / 32, HID / 32), tb, 0, stream>>>(W_out,   WoT,  HID, DIM);

  layernorm_kernel<<<MROWS / 8, 256, 0, stream>>>(x, ln_scale, ln_bias, normed);

  gemm_bf16<0><<<dim3(MROWS / 128, H2 / 128), 256, 0, stream>>>(
      normed, WhT, MROWS, H2, DIM, b_hidden, nullptr, nullptr, nullptr,
      vbuf, gbuf, nullptr);

  gemm_bf16<1><<<dim3(MROWS / 128, QKD / 128), 256, 0, stream>>>(
      normed, WqkT, MROWS, QKD, DIM, b_qk, gamma, beta, nullptr,
      qbuf, kbuf, nullptr);

  transpose_b2b<<<dim3(HID / 32, SEQ / 32, BATCH), tb, 0, stream>>>(
      vbuf, vTb, SEQ, HID);

  attention_kernel<<<dim3(SEQ / 128, HID / 256, BATCH), 256, 0, stream>>>(
      qbuf, kbuf, vTb, gbuf, Vout);

  gemm_bf16<2><<<dim3(MROWS / 128, DIM / 128), 256, 0, stream>>>(
      Vout, WoT, MROWS, DIM, HID, b_out, nullptr, nullptr, x,
      nullptr, nullptr, out);
}
